// AdversarialHead_9887014716223
// MI455X (gfx1250) — matrix operand order repaired and checked by reading, not compiled, not run on hardware
//
#include <hip/hip_runtime.h>
#include <hip/hip_bf16.h>

typedef __attribute__((ext_vector_type(16))) __bf16 v16bf;
typedef __attribute__((ext_vector_type(8)))  float  v8f;

union Frag {
  uint4  u[2];
  v16bf  v;
};

#define NFRAMES 2048
#define NAG     32
#define FEAT    512
#define HID     140
#define OUTN    256

// LDS strides (bf16 elements) chosen for bank-conflict-free 16-lane b128 loads
#define SA_STRIDE 552   // holds 544 used cols (512 feat + 18 one-hot + pad)
#define SN_STRIDE 520   // holds 512 used cols
#define SH_STRIDE 168   // holds 160 used cols (140 hidden + zero pad)

#define SA_BYTES (32 * SA_STRIDE * 2)   // 35328
#define SN_BYTES (32 * SN_STRIDE * 2)   // 33280
#define SH_BYTES (32 * SH_STRIDE * 2)   // 10752
#define SZ_BYTES (32 * 32 * 4)          // 4096
#define SMEM_BYTES (SA_BYTES + SN_BYTES + SH_BYTES + SZ_BYTES) // 83456

// ---- d_ws pre-swizzled bf16 weight-fragment layout (uint words) ----
// fragment = 32 lanes x 32 bytes (8 words/lane) of a 32x16 bf16 B tile
#define W1_FRAGS (9  * 17)   // N=144 (9 tiles) x K=544 (17 ksteps)
#define W2_FRAGS (16 * 5)    // N=256           x K=160
#define WI_FRAGS (2  * 32)   // N=32            x K=1024
#define W1_BASE  0
#define W2_BASE  (W1_FRAGS * 32 * 8)
#define WI_BASE  (W2_BASE + W2_FRAGS * 32 * 8)
#define TOTAL_SLOTS ((W1_FRAGS + W2_FRAGS + WI_FRAGS) * 32)  // 9504

// Output section offsets (flat floats, reference return order)
#define OFF_PRED 0
#define OFF_A0   16777216ULL   // 2048*32*256
#define OFF_A1   17301504ULL   // + 2048*32*8
#define OFF_A2   17694720ULL   // + 2048*32*6

__device__ __forceinline__ unsigned packbf(float a, float b) {
  __bf16 x = (__bf16)a, y = (__bf16)b;
  unsigned short ux = __builtin_bit_cast(unsigned short, x);
  unsigned short uy = __builtin_bit_cast(unsigned short, y);
  return (unsigned)ux | ((unsigned)uy << 16);
}

// B-fragment K index for VGPR v, lane-half h (32x16 B tile, bf16):
// VGPRs 0-3 pack K=8h+2v, +1; VGPRs 4-7 pack K=16+8h+2(v-4), +1
__device__ __forceinline__ int bfragK(int h, int v) { return h * 8 + 2 * v + (v >= 4 ? 8 : 0); }

// ---------------- Weight prep: fp32 -> bf16 pre-swizzled fragments ----------------
__global__ __launch_bounds__(256) void prep_weights(
    const float* __restrict__ W1, const float* __restrict__ W2,
    const float* __restrict__ Wi0, const float* __restrict__ Wi1,
    const float* __restrict__ Wi2, unsigned* __restrict__ ws) {
  int slot = blockIdx.x * blockDim.x + threadIdx.x;
  if (slot >= TOTAL_SLOTS) return;
  int frag = slot >> 5, lane = slot & 31;
  int h = lane >> 4, n16 = lane & 15;
  unsigned o[8];

  if (frag < W1_FRAGS) {                       // W1: 530x140 (K padded 544, N 144)
    int nt = frag / 17, ks = frag % 17;
    int n = nt * 16 + n16;
#pragma unroll
    for (int v = 0; v < 8; ++v) {
      int k0 = ks * 32 + bfragK(h, v);
      float f0 = (n < HID && k0     < 530) ? W1[(size_t)k0 * HID + n]       : 0.f;
      float f1 = (n < HID && k0 + 1 < 530) ? W1[(size_t)(k0 + 1) * HID + n] : 0.f;
      o[v] = packbf(f0, f1);
    }
  } else if (frag < W1_FRAGS + W2_FRAGS) {     // W2: 140x256 (K padded 160)
    int fr = frag - W1_FRAGS;
    int nt = fr / 5, ks = fr % 5;
    int n = nt * 16 + n16;
#pragma unroll
    for (int v = 0; v < 8; ++v) {
      int k0 = ks * 32 + bfragK(h, v);
      float f0 = (k0     < HID) ? W2[(size_t)k0 * OUTN + n]       : 0.f;
      float f1 = (k0 + 1 < HID) ? W2[(size_t)(k0 + 1) * OUTN + n] : 0.f;
      o[v] = packbf(f0, f1);
    }
  } else {                                     // Wi concat: 1024 x 18 (N padded 32)
    int fr = frag - W1_FRAGS - W2_FRAGS;
    int nt = fr / 32, ks = fr % 32;
    int c = nt * 16 + n16;
#pragma unroll
    for (int v = 0; v < 8; ++v) {
      int k0 = ks * 32 + bfragK(h, v);
      float f0 = 0.f, f1 = 0.f;
      if (c < 8)       { f0 = Wi0[(size_t)k0 * 8 + c];        f1 = Wi0[(size_t)(k0 + 1) * 8 + c]; }
      else if (c < 14) { f0 = Wi1[(size_t)k0 * 6 + (c - 8)];  f1 = Wi1[(size_t)(k0 + 1) * 6 + (c - 8)]; }
      else if (c < 18) { f0 = Wi2[(size_t)k0 * 4 + (c - 14)]; f1 = Wi2[(size_t)(k0 + 1) * 4 + (c - 14)]; }
      o[v] = packbf(f0, f1);
    }
  }
  uint4* dst = (uint4*)(ws + (size_t)slot * 8);
  dst[0] = make_uint4(o[0], o[1], o[2], o[3]);
  dst[1] = make_uint4(o[4], o[5], o[6], o[7]);
}

// ---------------- Main fused kernel: one block per frame ----------------
__global__ __launch_bounds__(256) void icm_kernel(
    const float* __restrict__ curr, const float* __restrict__ nxt,
    const int* __restrict__ actions,
    const float* __restrict__ b1, const float* __restrict__ b2,
    const float* __restrict__ bi0, const float* __restrict__ bi1,
    const float* __restrict__ bi2,
    const unsigned* __restrict__ ws, float* __restrict__ out) {
  extern __shared__ __align__(16) char smem[];
  __bf16* sA = (__bf16*)smem;                               // [32][SA_STRIDE]
  __bf16* sN = (__bf16*)(smem + SA_BYTES);                  // [32][SN_STRIDE]
  __bf16* sH = (__bf16*)(smem + SA_BYTES + SN_BYTES);       // [32][SH_STRIDE]
  float*  sZ = (float*)(smem + SA_BYTES + SN_BYTES + SH_BYTES); // [32][32]

  const int f = blockIdx.x;
  const int tid = threadIdx.x, wave = tid >> 5, lane = tid & 31;
  const int h = lane >> 4, m16 = lane & 15;

  // --- Stage 0: features -> LDS bf16; one-hot cols; zero pads ---
  const float4* gc = (const float4*)(curr + (size_t)f * NAG * FEAT);
  const float4* gn = (const float4*)(nxt  + (size_t)f * NAG * FEAT);
  for (int i = tid; i < NAG * FEAT / 4; i += 256) {   // 4096 float4 per tensor
    int row = i >> 7, c4 = (i & 127) * 4;
    float4 vc = gc[i];
    __bf16* d = sA + row * SA_STRIDE + c4;
    d[0] = (__bf16)vc.x; d[1] = (__bf16)vc.y; d[2] = (__bf16)vc.z; d[3] = (__bf16)vc.w;
    float4 vn = gn[i];
    __bf16* e = sN + row * SN_STRIDE + c4;
    e[0] = (__bf16)vn.x; e[1] = (__bf16)vn.y; e[2] = (__bf16)vn.z; e[3] = (__bf16)vn.w;
  }
  for (int i = tid; i < 32 * 32; i += 256) {          // zero one-hot region 512..543
    int row = i >> 5, c = i & 31;
    sA[row * SA_STRIDE + 512 + c] = (__bf16)0.f;
  }
  for (int i = tid; i < 32 * 20; i += 256) {          // zero hidden K-pad 140..159
    int row = i / 20, c = 140 + i % 20;
    sH[row * SH_STRIDE + c] = (__bf16)0.f;
  }
  __syncthreads();
  if (tid < 96) {                                     // set one-hot ones
    int t = tid >> 5, a = tid & 31;
    int act = actions[(size_t)f * 96 + t * 32 + a];
    int st = (t == 0) ? 0 : (t == 1) ? 8 : 14;
    sA[a * SA_STRIDE + 512 + st + act] = (__bf16)1.0f;
  }
  __syncthreads();

  // --- Stage 1: hidden = fwd_in @ W1 + b1  (18 MN tiles over 8 waves) ---
  for (int t = wave; t < 18; t += 8) {
    int mt = t & 1, nt = t >> 1;
    v8f acc = {};
    int row = mt * 16 + m16;
    const char* aBase = (const char*)sA + (size_t)row * (SA_STRIDE * 2);
#pragma unroll 4
    for (int ks = 0; ks < 17; ++ks) {
      Frag fa, fb;
      int kb = ks * 64;
      fa.u[0] = *(const uint4*)(aBase + kb + h * 16);
      fa.u[1] = *(const uint4*)(aBase + kb + 32 + h * 16);
      const uint4* bp = (const uint4*)(ws + W1_BASE + (size_t)((nt * 17 + ks) * 32 + lane) * 8);
      fb.u[0] = bp[0]; fb.u[1] = bp[1];
      acc = __builtin_amdgcn_wmma_f32_16x16x32_bf16(false, fa.v, false, fb.v,
                                                    (short)0, acc, false, false);
    }
    int col = nt * 16 + m16;
    float bias = (col < HID) ? b1[col] : 0.f;
    if (col < HID) {
#pragma unroll
      for (int r = 0; r < 8; ++r) {
        int orow = mt * 16 + r + 8 * h;
        sH[orow * SH_STRIDE + col] = (__bf16)(acc[r] + bias);
      }
    }
  }
  __syncthreads();

  // --- Stage 2: pred = hidden @ W2 + b2  (32 MN tiles over 8 waves) ---
  for (int t = wave; t < 32; t += 8) {
    int mt = t & 1, nt = t >> 1;
    v8f acc = {};
    int row = mt * 16 + m16;
    const char* aBase = (const char*)sH + (size_t)row * (SH_STRIDE * 2);
#pragma unroll
    for (int ks = 0; ks < 5; ++ks) {
      Frag fa, fb;
      int kb = ks * 64;
      fa.u[0] = *(const uint4*)(aBase + kb + h * 16);
      fa.u[1] = *(const uint4*)(aBase + kb + 32 + h * 16);
      const uint4* bp = (const uint4*)(ws + W2_BASE + (size_t)((nt * 5 + ks) * 32 + lane) * 8);
      fb.u[0] = bp[0]; fb.u[1] = bp[1];
      acc = __builtin_amdgcn_wmma_f32_16x16x32_bf16(false, fa.v, false, fb.v,
                                                    (short)0, acc, false, false);
    }
    int col = nt * 16 + m16;
    float bias = b2[col];
    float* o = out + OFF_PRED + (size_t)f * (NAG * OUTN) + col;
#pragma unroll
    for (int r = 0; r < 8; ++r) {
      int orow = mt * 16 + r + 8 * h;
      o[(size_t)orow * OUTN] = acc[r] + bias;
    }
  }

  // --- Stage 3: inverse logits = [curr|next] @ Wi + bi  (4 tiles, waves 0..3) ---
  if (wave < 4) {
    int mt = wave & 1, nt = wave >> 1;
    v8f acc = {};
    int row = mt * 16 + m16;
    const char* aC = (const char*)sA + (size_t)row * (SA_STRIDE * 2);
    const char* aN = (const char*)sN + (size_t)row * (SN_STRIDE * 2);
#pragma unroll 4
    for (int ks = 0; ks < 32; ++ks) {
      const char* aBase = (ks < 16) ? aC : aN;
      int kb = ((ks < 16) ? ks : (ks - 16)) * 64;
      Frag fa, fb;
      fa.u[0] = *(const uint4*)(aBase + kb + h * 16);
      fa.u[1] = *(const uint4*)(aBase + kb + 32 + h * 16);
      const uint4* bp = (const uint4*)(ws + WI_BASE + (size_t)((nt * 32 + ks) * 32 + lane) * 8);
      fb.u[0] = bp[0]; fb.u[1] = bp[1];
      acc = __builtin_amdgcn_wmma_f32_16x16x32_bf16(false, fa.v, false, fb.v,
                                                    (short)0, acc, false, false);
    }
    int col = nt * 16 + m16;
    float bias = (col < 8) ? bi0[col] : (col < 14) ? bi1[col - 8]
               : (col < 18) ? bi2[col - 14] : 0.f;
#pragma unroll
    for (int r = 0; r < 8; ++r) {
      int orow = mt * 16 + r + 8 * h;
      sZ[orow * 32 + col] = acc[r] + bias;
    }
  }
  __syncthreads();

  // --- Stage 4: softmax over the 32 agents, per logit column ---
  if (tid < 18) {
    int c = tid;
    float m = -3.4e38f;
#pragma unroll
    for (int r = 0; r < 32; ++r) m = fmaxf(m, sZ[r * 32 + c]);
    float e[32], s = 0.f;
#pragma unroll
    for (int r = 0; r < 32; ++r) { e[r] = __expf(sZ[r * 32 + c] - m); s += e[r]; }
    float inv = 1.f / s;
    size_t base; int stride;
    if (c < 8)       { base = OFF_A0 + (size_t)f * 256 + c;        stride = 8; }
    else if (c < 14) { base = OFF_A1 + (size_t)f * 192 + (c - 8);  stride = 6; }
    else             { base = OFF_A2 + (size_t)f * 128 + (c - 14); stride = 4; }
#pragma unroll
    for (int r = 0; r < 32; ++r) out[base + (size_t)r * stride] = e[r] * inv;
  }
}

extern "C" void kernel_launch(void* const* d_in, const int* in_sizes, int n_in,
                              void* d_out, int out_size, void* d_ws, size_t ws_size,
                              hipStream_t stream) {
  const float* curr = (const float*)d_in[0];
  const float* nxt  = (const float*)d_in[1];
  const int*   act  = (const int*)d_in[2];
  const float* W1   = (const float*)d_in[3];
  const float* b1   = (const float*)d_in[4];
  const float* W2   = (const float*)d_in[5];
  const float* b2   = (const float*)d_in[6];
  const float* Wi0  = (const float*)d_in[7];
  const float* bi0  = (const float*)d_in[8];
  const float* Wi1  = (const float*)d_in[9];
  const float* bi1  = (const float*)d_in[10];
  const float* Wi2  = (const float*)d_in[11];
  const float* bi2  = (const float*)d_in[12];
  unsigned* ws = (unsigned*)d_ws;
  float* out = (float*)d_out;

  (void)hipFuncSetAttribute((const void*)icm_kernel,
                            hipFuncAttributeMaxDynamicSharedMemorySize, SMEM_BYTES);

  prep_weights<<<(TOTAL_SLOTS + 255) / 256, 256, 0, stream>>>(W1, W2, Wi0, Wi1, Wi2, ws);
  icm_kernel<<<NFRAMES, 256, SMEM_BYTES, stream>>>(curr, nxt, act, b1, b2,
                                                   bi0, bi1, bi2, ws, out);
}
